// MultiHeadedAttention_62380105007823
// MI455X (gfx1250) — matrix operand order repaired and checked by reading, not compiled, not run on hardware
//
#include <hip/hip_runtime.h>
#include <hip/hip_bf16.h>

typedef __bf16 bf16;
typedef __attribute__((ext_vector_type(16))) bf16  v16bf;
typedef __attribute__((ext_vector_type(8)))  float v8f;
typedef __attribute__((ext_vector_type(4)))  float f32x4;
typedef __attribute__((ext_vector_type(4)))  int   i32x4;

constexpr int Bb = 16, Nn = 512, Dd = 256, Hh = 8, DK = 32;
constexpr float SCALE  = 0.17677669529663687f;   // 1/sqrt(DK)
constexpr float L_ATT  = 0.3f, L_DIST = 0.3f, L_ADJ = 0.4f;
constexpr float EPSv   = 1e-6f;
constexpr float NEGBIG = -1.0e12f;

static __device__ __forceinline__ v8f wmma_bf16(v16bf a, v16bf b, v8f c) {
  // D = A(16x32 bf16) * B(32x16 bf16) + C(16x16 f32)
  return __builtin_amdgcn_wmma_f32_16x16x32_bf16(false, a, false, b,
                                                 (short)0, c, false, false);
}

// ---------------------------------------------------------------------------
// Stage 0: per-row stats for the dist-softmax and adjacency normalization.
// One wave per (b, query-row). Also pulls adj/dist into L2 (they fit: 33.5 MB).
// ---------------------------------------------------------------------------
__global__ __launch_bounds__(32) void rowstats_kernel(
    const float* __restrict__ adj, const float* __restrict__ dist,
    const int* __restrict__ mask,
    float* __restrict__ mD, float* __restrict__ cD, float* __restrict__ iA) {
  const int row  = blockIdx.x;          // 0 .. B*N-1
  const int b    = row >> 9;
  const int lane = threadIdx.x;
  const float* dp = dist + (size_t)row * Nn;
  const float* gp = adj  + (size_t)row * Nn;
  const int*   mp = mask + (size_t)b * Nn;

  float dv[16], av[16]; int mv[16];
#pragma unroll
  for (int i = 0; i < 16; i++) {
    const int idx = lane + i * 32;
    dv[i] = dp[idx]; av[i] = gp[idx]; mv[i] = mp[idx];
  }
  float mx = -3.0e38f, asum = 0.f;
#pragma unroll
  for (int i = 0; i < 16; i++) {
    asum += av[i];
    if (mv[i]) mx = fmaxf(mx, -dv[i]);
  }
#pragma unroll
  for (int sx = 1; sx < 32; sx <<= 1) {
    mx = fmaxf(mx, __shfl_xor(mx, sx, 32));
    asum += __shfl_xor(asum, sx, 32);
  }
  float s = 0.f;
#pragma unroll
  for (int i = 0; i < 16; i++)
    if (mv[i]) s += __expf(-dv[i] - mx);
#pragma unroll
  for (int sx = 1; sx < 32; sx <<= 1) s += __shfl_xor(s, sx, 32);

  if (lane == 0) {
    mD[row] = mx;
    cD[row] = L_DIST / s;               // 0.3 * softmax(-d) = cD * exp(-d - mD)
    iA[row] = L_ADJ / (asum + EPSv);    // 0.4 * adj / (rowsum + eps)
  }
}

// ---------------------------------------------------------------------------
// Stage 1: Q/K/V projections, bf16 WMMA. One wave computes a 16x64 tile.
// Q is pre-scaled by 1/sqrt(DK). V is stored transposed [B,H,DK,N].
// ---------------------------------------------------------------------------
__global__ __launch_bounds__(32) void qkv_proj_kernel(
    const float* __restrict__ qin, const float* __restrict__ kin,
    const float* __restrict__ vin,
    const float* __restrict__ Wq, const float* __restrict__ bq,
    const float* __restrict__ Wk, const float* __restrict__ bk,
    const float* __restrict__ Wv, const float* __restrict__ bv,
    bf16* __restrict__ Qbf, bf16* __restrict__ Kbf, bf16* __restrict__ Vt) {
  const int mt = blockIdx.x, nb = blockIdx.y, z = blockIdx.z;
  const float* X    = (z == 0) ? qin : (z == 1) ? kin : vin;
  const float* W    = (z == 0) ? Wq  : (z == 1) ? Wk  : Wv;
  const float* bias = (z == 0) ? bq  : (z == 1) ? bk  : bv;

  const int lane = threadIdx.x, l15 = lane & 15, half = lane >> 4;
  const int arow  = mt * 16 + l15;
  const int kbA   = half ? 16 : 0;   // A-layout K base per half-wave
  const int koffB = half ? 16 : 0;   // B-layout K base per half-wave

  const v8f zed = {0.f, 0.f, 0.f, 0.f, 0.f, 0.f, 0.f, 0.f};
  v8f acc[4];
#pragma unroll
  for (int j = 0; j < 4; j++) acc[j] = zed;

  for (int k0 = 0; k0 < Dd; k0 += 32) {
    const float* apx = X + (size_t)arow * Dd + k0 + kbA;
    f32x4 a0 = *(const f32x4*)(apx),      a1 = *(const f32x4*)(apx + 4),
          a2 = *(const f32x4*)(apx + 8), a3 = *(const f32x4*)(apx + 12);
    v16bf A;
#pragma unroll
    for (int i = 0; i < 4; i++) {
      A[i]      = (bf16)a0[i]; A[4 + i]  = (bf16)a1[i];
      A[8 + i]  = (bf16)a2[i]; A[12 + i] = (bf16)a3[i];
    }
#pragma unroll
    for (int j = 0; j < 4; j++) {
      const int ncol = nb * 64 + j * 16 + l15;   // Y = X @ W^T -> B[k,n] = W[n,k]
      const float* wp = W + (size_t)ncol * Dd + k0 + koffB;
      f32x4 w0 = *(const f32x4*)(wp),     w1 = *(const f32x4*)(wp + 4),
            w2 = *(const f32x4*)(wp + 8), w3 = *(const f32x4*)(wp + 12);
      v16bf Bm;
#pragma unroll
      for (int i = 0; i < 4; i++) {
        Bm[i]      = (bf16)w0[i]; Bm[4 + i]  = (bf16)w1[i];
        Bm[8 + i]  = (bf16)w2[i]; Bm[12 + i] = (bf16)w3[i];
      }
      acc[j] = wmma_bf16(A, Bm, acc[j]);
    }
  }

#pragma unroll
  for (int j = 0; j < 4; j++) {
    const int ncol = nb * 64 + j * 16 + l15;
    const float bv_ = bias[ncol];
    const int hh = ncol >> 5, dk = ncol & 31;
#pragma unroll
    for (int r = 0; r < 8; r++) {
      const int m   = mt * 16 + r + (half ? 8 : 0);   // C-layout row
      const int bi  = m >> 9, tok = m & (Nn - 1);
      float y = acc[j][r] + bv_;
      if (z == 0) y *= SCALE;
      const bf16 yb = (bf16)y;
      if (z == 2)
        Vt[((size_t)(bi * Hh + hh) * DK + dk) * Nn + tok] = yb;
      else if (z == 0)
        Qbf[((size_t)(bi * Hh + hh) * Nn + tok) * DK + dk] = yb;
      else
        Kbf[((size_t)(bi * Hh + hh) * Nn + tok) * DK + dk] = yb;
    }
  }
}

// ---------------------------------------------------------------------------
// Stage 2: fused attention. One wave per (b, h, 16-query tile).
// Online softmax for the attention term; dist/adj term accumulated separately
// (it is already normalized) so both feed WMMA against shared V operands.
// ---------------------------------------------------------------------------
__global__ __launch_bounds__(32) void attn_kernel(
    const bf16* __restrict__ Qbf, const bf16* __restrict__ Kbf,
    const bf16* __restrict__ Vt,
    const float* __restrict__ adj, const float* __restrict__ dist,
    const int* __restrict__ mask,
    const float* __restrict__ mD, const float* __restrict__ cD,
    const float* __restrict__ iA, bf16* __restrict__ Xbf) {
  __shared__ bf16 pL[16 * 32];   // C-layout -> A-layout bounce buffer for P

  const int qt = blockIdx.x, h = blockIdx.y, b = blockIdx.z;
  const int lane = threadIdx.x, l15 = lane & 15, half = lane >> 4;
  const int q0 = qt * 16;
  const int kbA = half ? 16 : 0, koffB = half ? 16 : 0;
  const size_t bh = (size_t)(b * Hh + h);

  // Persistent Q A-operand (row = l15, DK = 32 -> exactly one WMMA per S tile)
  const int qrow = q0 + l15;
  const bf16* qp = Qbf + (bh * Nn + qrow) * DK;
  v16bf Aq;
#pragma unroll
  for (int i = 0; i < 8; i++) { Aq[i] = qp[kbA + i]; Aq[8 + i] = qp[kbA + 8 + i]; }

  const float mDq = mD[b * Nn + qrow];
  const float cDq = cD[b * Nn + qrow];
  const float iAq = iA[b * Nn + qrow];

  const v8f zed = {0.f, 0.f, 0.f, 0.f, 0.f, 0.f, 0.f, 0.f};
  v8f o0 = zed, o1 = zed, e0 = zed, e1 = zed;
  float mrun[8], lrun[8];
#pragma unroll
  for (int r = 0; r < 8; r++) { mrun[r] = -3.0e38f; lrun[r] = 0.f; }

  const float* dbase = dist + (size_t)b * Nn * Nn + (size_t)qrow * Nn;
  const float* gbase = adj  + (size_t)b * Nn * Nn + (size_t)qrow * Nn;
  const int*   mbase = mask + (size_t)b * Nn;

  for (int kc = 0; kc < Nn; kc += 32) {
    // ---- S = Qs * K^T (two 16x16 key tiles, one WMMA each) ----
    const bf16* kp0 = Kbf + (bh * Nn + kc + l15) * DK + koffB;
    const bf16* kp1 = kp0 + 16 * DK;
    v16bf Bk0 = *(const v16bf*)kp0;
    v16bf Bk1 = *(const v16bf*)kp1;
    v8f s0 = wmma_bf16(Aq, Bk0, zed);
    v8f s1 = wmma_bf16(Aq, Bk1, zed);

    const bool mk0 = mbase[kc + l15] != 0;
    const bool mk1 = mbase[kc + 16 + l15] != 0;

    // ---- online softmax update (row lives across 16 lanes of a half) ----
#pragma unroll
    for (int r = 0; r < 8; r++) {
      float a0 = mk0 ? s0[r] : NEGBIG;
      float a1 = mk1 ? s1[r] : NEGBIG;
      float v = fmaxf(a0, a1);
#pragma unroll
      for (int sx = 1; sx < 16; sx <<= 1) v = fmaxf(v, __shfl_xor(v, sx, 32));
      const float mnew  = fmaxf(mrun[r], v);
      const float alpha = __expf(mrun[r] - mnew);
      const float p0 = __expf(a0 - mnew), p1 = __expf(a1 - mnew);
      float rs = p0 + p1;
#pragma unroll
      for (int sx = 1; sx < 16; sx <<= 1) rs += __shfl_xor(rs, sx, 32);
      lrun[r] = lrun[r] * alpha + rs;
      mrun[r] = mnew;
      o0[r] *= alpha; o1[r] *= alpha;
      const int prow = r + (half ? 8 : 0);
      pL[prow * 32 + l15]      = (bf16)p0;
      pL[prow * 32 + 16 + l15] = (bf16)p1;
    }
    asm volatile("s_wait_dscnt 0" ::: "memory");

    // Re-read P in A-operand layout from LDS
    v16bf Ap;
    const bf16* pr = pL + l15 * 32 + kbA;
#pragma unroll
    for (int i = 0; i < 8; i++) { Ap[i] = pr[i]; Ap[8 + i] = pr[8 + i]; }

    // ---- V B-operands (Vt is [B,H,DK,N] so loads are contiguous) ----
    const bf16* vp0 = Vt + (bh * DK + l15) * Nn + kc + koffB;
    const bf16* vp1 = Vt + (bh * DK + 16 + l15) * Nn + kc + koffB;
    v16bf Bv0 = *(const v16bf*)vp0;
    v16bf Bv1 = *(const v16bf*)vp1;

    o0 = wmma_bf16(Ap, Bv0, o0);
    o1 = wmma_bf16(Ap, Bv1, o1);

    // ---- pre-normalized blend term 0.3*p_dist + 0.4*p_adj, in A-layout ----
    const float* dp = dbase + kc + kbA;
    const float* gp = gbase + kc + kbA;
    const int*   mp = mbase + kc + kbA;
    f32x4 dd0 = *(const f32x4*)(dp),      dd1 = *(const f32x4*)(dp + 4),
          dd2 = *(const f32x4*)(dp + 8), dd3 = *(const f32x4*)(dp + 12);
    f32x4 gg0 = *(const f32x4*)(gp),      gg1 = *(const f32x4*)(gp + 4),
          gg2 = *(const f32x4*)(gp + 8), gg3 = *(const f32x4*)(gp + 12);
    i32x4 mm0 = *(const i32x4*)(mp),      mm1 = *(const i32x4*)(mp + 4),
          mm2 = *(const i32x4*)(mp + 8), mm3 = *(const i32x4*)(mp + 12);
    v16bf Ae;
#pragma unroll
    for (int i = 0; i < 4; i++) {
      Ae[i]      = (bf16)(iAq * gg0[i] + (mm0[i] ? cDq * __expf(-dd0[i] - mDq) : 0.f));
      Ae[4 + i]  = (bf16)(iAq * gg1[i] + (mm1[i] ? cDq * __expf(-dd1[i] - mDq) : 0.f));
      Ae[8 + i]  = (bf16)(iAq * gg2[i] + (mm2[i] ? cDq * __expf(-dd2[i] - mDq) : 0.f));
      Ae[12 + i] = (bf16)(iAq * gg3[i] + (mm3[i] ? cDq * __expf(-dd3[i] - mDq) : 0.f));
    }
    e0 = wmma_bf16(Ae, Bv0, e0);
    e1 = wmma_bf16(Ae, Bv1, e1);
  }

  // ---- finalize: 0.3 * softmax-part / l + blend-part; store [B,N,D] bf16 ----
#pragma unroll
  for (int r = 0; r < 8; r++) {
    const float invl = L_ATT / lrun[r];
    const int tok = q0 + r + (half ? 8 : 0);
    const size_t obase = ((size_t)b * Nn + tok) * Dd + h * DK;
    Xbf[obase + l15]      = (bf16)(o0[r] * invl + e0[r]);
    Xbf[obase + 16 + l15] = (bf16)(o1[r] * invl + e1[r]);
  }
}

// ---------------------------------------------------------------------------
// Stage 3: output projection out = X @ Wo^T + bo (f32 output).
// ---------------------------------------------------------------------------
__global__ __launch_bounds__(32) void out_proj_kernel(
    const bf16* __restrict__ Xbf, const float* __restrict__ Wo,
    const float* __restrict__ bo, float* __restrict__ out) {
  const int mt = blockIdx.x, nb = blockIdx.y;
  const int lane = threadIdx.x, l15 = lane & 15, half = lane >> 4;
  const int kbA = half ? 16 : 0, koffB = half ? 16 : 0;
  const int arow = mt * 16 + l15;

  const v8f zed = {0.f, 0.f, 0.f, 0.f, 0.f, 0.f, 0.f, 0.f};
  v8f acc[4];
#pragma unroll
  for (int j = 0; j < 4; j++) acc[j] = zed;

  for (int k0 = 0; k0 < Dd; k0 += 32) {
    const bf16* xp = Xbf + (size_t)arow * Dd + k0 + kbA;
    v16bf A;
#pragma unroll
    for (int i = 0; i < 8; i++) { A[i] = xp[i]; A[8 + i] = xp[8 + i]; }
#pragma unroll
    for (int j = 0; j < 4; j++) {
      const int ncol = nb * 64 + j * 16 + l15;
      const float* wp = Wo + (size_t)ncol * Dd + k0 + koffB;
      f32x4 w0 = *(const f32x4*)(wp),     w1 = *(const f32x4*)(wp + 4),
            w2 = *(const f32x4*)(wp + 8), w3 = *(const f32x4*)(wp + 12);
      v16bf Bm;
#pragma unroll
      for (int i = 0; i < 4; i++) {
        Bm[i]      = (bf16)w0[i]; Bm[4 + i]  = (bf16)w1[i];
        Bm[8 + i]  = (bf16)w2[i]; Bm[12 + i] = (bf16)w3[i];
      }
      acc[j] = wmma_bf16(A, Bm, acc[j]);
    }
  }
#pragma unroll
  for (int j = 0; j < 4; j++) {
    const int ncol = nb * 64 + j * 16 + l15;
    const float bb = bo[ncol];
#pragma unroll
    for (int r = 0; r < 8; r++) {
      const int m = mt * 16 + r + (half ? 8 : 0);
      out[(size_t)m * Dd + ncol] = acc[j][r] + bb;
    }
  }
}

// ---------------------------------------------------------------------------
extern "C" void kernel_launch(void* const* d_in, const int* in_sizes, int n_in,
                              void* d_out, int out_size, void* d_ws,
                              size_t ws_size, hipStream_t stream) {
  (void)in_sizes; (void)n_in; (void)out_size; (void)ws_size;
  const float* query = (const float*)d_in[0];
  const float* key_  = (const float*)d_in[1];
  const float* value = (const float*)d_in[2];
  const float* adj   = (const float*)d_in[3];
  const float* dist  = (const float*)d_in[4];
  /* d_in[5] edges_att: unused by reference */
  const int*   mask  = (const int*)d_in[6];
  const float* Wq = (const float*)d_in[7],  *bq = (const float*)d_in[8];
  const float* Wk = (const float*)d_in[9],  *bk = (const float*)d_in[10];
  const float* Wv = (const float*)d_in[11], *bv = (const float*)d_in[12];
  const float* Wo = (const float*)d_in[13], *bo = (const float*)d_in[14];
  float* out = (float*)d_out;

  char* ws = (char*)d_ws;
  const size_t szQKV = (size_t)Bb * Hh * Nn * DK * sizeof(bf16);  // 4 MiB
  const size_t szX   = (size_t)Bb * Nn * Dd * sizeof(bf16);       // 4 MiB
  bf16*  Qbf = (bf16*)(ws);
  bf16*  Kbf = (bf16*)(ws + szQKV);
  bf16*  Vt  = (bf16*)(ws + 2 * szQKV);
  bf16*  Xbf = (bf16*)(ws + 3 * szQKV);
  float* mD  = (float*)(ws + 3 * szQKV + szX);
  float* cD  = mD + Bb * Nn;
  float* iA  = cD + Bb * Nn;

  rowstats_kernel<<<dim3(Bb * Nn), 32, 0, stream>>>(adj, dist, mask, mD, cD, iA);
  qkv_proj_kernel<<<dim3((Bb * Nn) / 16, Dd / 64, 3), 32, 0, stream>>>(
      query, key_, value, Wq, bq, Wk, bk, Wv, bv, Qbf, Kbf, Vt);
  attn_kernel<<<dim3(Nn / 16, Hh, Bb), 32, 0, stream>>>(
      Qbf, Kbf, Vt, adj, dist, mask, mD, cD, iA, Xbf);
  out_proj_kernel<<<dim3((Bb * Nn) / 16, Dd / 64), 32, 0, stream>>>(Xbf, Wo, bo, out);
}
